// MutiHeadSelfAttention_48404281426511
// MI455X (gfx1250) — matrix operand order repaired and checked by reading, not compiled, not run on hardware
//
#include <hip/hip_runtime.h>

typedef unsigned short u16;
typedef unsigned int   u32;
typedef __attribute__((ext_vector_type(16))) __bf16 v16bf;
typedef __attribute__((ext_vector_type(8)))  float  v8f;
typedef __attribute__((__vector_size__(4 * sizeof(int)))) int v4i;

#define NB   2
#define TSEQ 2048
#define DMOD 1024
#define NH   16
#define DKH  64
#define BT   (NB * TSEQ)        // 4096 rows
#define E3   (3 * DMOD)         // 3072

#if defined(__has_builtin)
#if __has_builtin(__builtin_amdgcn_global_load_async_to_lds_b128) && \
    __has_builtin(__builtin_amdgcn_s_wait_asynccnt)
#define USE_ASYNC_LDS 1
#endif
#endif
#ifndef USE_ASYNC_LDS
#define USE_ASYNC_LDS 0
#endif

union Frag16 { u32 u[8]; v16bf v; };

// ---------- fragment index helpers (cdna5 CDNA5 ISA 7.12.2) ----------
__device__ __forceinline__ int akp(int v, int half) {
  return (v < 4 ? 2 * v : 8 + 2 * v) + 8 * half;   // A 16x32: K-pair index
}
__device__ __forceinline__ int bkp(int v, int half) { return (v < 4 ? 2 * v : 8 + 2 * v) + 8 * half; }

__device__ __forceinline__ u32 ld32(const u16* p) {
  return *reinterpret_cast<const u32*>(p);
}

__device__ __forceinline__ u16 f2bf(float f) {
  u32 u = __float_as_uint(f);
  u32 r = u + 0x7FFFu + ((u >> 16) & 1u);
  return (u16)(r >> 16);
}

__device__ __forceinline__ v8f wmma_bf16(const Frag16& a, const Frag16& b, v8f c) {
  return __builtin_amdgcn_wmma_f32_16x16x32_bf16(false, a.v, false, b.v,
                                                 (short)0, c, false, false);
}

// copy one 16-byte chunk global -> LDS (async path if toolchain exposes it)
__device__ __forceinline__ void stage16(const u16* g, u16* l) {
#if USE_ASYNC_LDS
  __builtin_amdgcn_global_load_async_to_lds_b128((v4i*)g, (v4i*)l, 0, 0);
#else
  *reinterpret_cast<uint4*>(l) = *reinterpret_cast<const uint4*>(g);
#endif
}
__device__ __forceinline__ void stage_wait() {
#if USE_ASYNC_LDS
  __builtin_amdgcn_s_wait_asynccnt(0);
#endif
}

// ---------------- f32 -> bf16 conversion ----------------
__global__ void cvt_bf16(const float* __restrict__ in, u16* __restrict__ out, int n) {
  int i = blockIdx.x * blockDim.x + threadIdx.x;
  if (i < n) out[i] = f2bf(in[i]);
}

// ---------------- shared GEMM core ----------------
// C[32x64 per wave] = A[32 x K] * W[64 x K]^T. block = 8 waves, N=512/block.
// A tile (32 x 128) staged through LDS per barrier period.
__device__ __forceinline__ void gemm_core(const u16* __restrict__ A,
                                          const u16* __restrict__ W,
                                          int mbase, int ebase, int K,
                                          u16* atile, v8f c[2][4]) {
  const int lane = threadIdx.x & 31;
  const int half = lane >> 4, l16 = lane & 15;
#pragma unroll
  for (int mt = 0; mt < 2; ++mt)
#pragma unroll
    for (int nt = 0; nt < 4; ++nt) c[mt][nt] = (v8f){0,0,0,0,0,0,0,0};

  for (int kc = 0; kc < K; kc += 128) {
    __syncthreads();           // previous period fully consumed
#pragma unroll
    for (int i = 0; i < 2; ++i) {      // 32 rows x 128 cols bf16 = 8KB
      int idx = threadIdx.x + i * 256;
      int row = idx >> 4, ck = idx & 15;
      stage16(A + (size_t)(mbase + row) * K + kc + ck * 8,
              atile + row * 128 + ck * 8);
    }
    stage_wait();              // per-wave ASYNCcnt drain before signaling
    __syncthreads();

#pragma unroll
    for (int ks = 0; ks < 4; ++ks) {
      Frag16 a[2];
#pragma unroll
      for (int mt = 0; mt < 2; ++mt)
#pragma unroll
        for (int v = 0; v < 8; ++v)
          a[mt].u[v] = *reinterpret_cast<const u32*>(
              atile + (mt * 16 + l16) * 128 + ks * 32 + akp(v, half));
#pragma unroll
      for (int nt = 0; nt < 4; ++nt) {
        Frag16 b;
        const u16* wr = W + (size_t)(ebase + nt * 16 + l16) * K + kc + ks * 32;
#pragma unroll
        for (int v = 0; v < 8; ++v) b.u[v] = ld32(wr + bkp(v, half));
#pragma unroll
        for (int mt = 0; mt < 2; ++mt) c[mt][nt] = wmma_bf16(a[mt], b, c[mt][nt]);
      }
    }
  }
}

// ---------------- QKV projection + RoPE + layout ----------------
__global__ void qkv_rope_kernel(const u16* __restrict__ xb, const u16* __restrict__ wb,
                                u16* __restrict__ qb, u16* __restrict__ kb,
                                u16* __restrict__ vtb) {
  __shared__ u16 atile[32 * 128];
  const int lane = threadIdx.x & 31, wave = threadIdx.x >> 5;
  const int half = lane >> 4, l16 = lane & 15;
  const int mbase = blockIdx.y * 32;
  const int ebase = blockIdx.x * 512 + wave * 64;

  v8f c[2][4];
  gemm_core(xb, wb, mbase, ebase, DMOD, atile, c);

  const float LOG_TH_64 = 0.14391565f;   // ln(10000)/64
#pragma unroll
  for (int mt = 0; mt < 2; ++mt)
#pragma unroll
    for (int nt = 0; nt < 4; ++nt) {
#pragma unroll
      for (int r = 0; r < 8; ++r) {
        float val = c[mt][nt][r];
        float partner = __shfl_xor(val, 1);
        int e = ebase + nt * 16 + l16;
        int m = mbase + mt * 16 + r + 8 * half;
        int sec = e >> 10, hd = e & 1023, h = hd >> 6, d = hd & 63;
        int bidx = m >> 11, t = m & (TSEQ - 1);
        if (sec < 2) { // RoPE on q,k
          float inv = __expf(-(float)(d & ~1) * LOG_TH_64);
          float ang = (float)t * inv;
          float cs = __cosf(ang), sn = __sinf(ang);
          float rot = (d & 1) ? partner : -partner;
          val = val * cs + rot * sn;
          if (sec == 0) val *= 0.125f;   // fold 1/sqrt(dk)
        }
        int bh = bidx * NH + h;
        u16 o = f2bf(val);
        if (sec == 0)      qb[((size_t)bh * TSEQ + t) * DKH + d] = o;
        else if (sec == 1) kb[((size_t)bh * TSEQ + t) * DKH + d] = o;
        else               vtb[((size_t)bh * DKH + d) * TSEQ + t] = o;
      }
    }
}

// ---------------- flash attention (one wave per 32-row q tile) ----------------
__global__ void attn_kernel(const u16* __restrict__ qb, const u16* __restrict__ kbm,
                            const u16* __restrict__ vtb, u16* __restrict__ attnb) {
  __shared__ u16 plds[8][2 * 16 * 32];     // two 16x32 P tiles per wave
  const int lane = threadIdx.x & 31, wave = threadIdx.x >> 5;
  const int half = lane >> 4, l16 = lane & 15;
  const int wid = blockIdx.x * 8 + wave;
  const int bh = wid >> 6, qtile = wid & 63;      // 64 tiles of 32 rows
  const int qbase = qtile * 32;
  const int b = bh >> 4, h = bh & 15;

  // Q A-fragments for both 16-row tiles (already scaled by 1/8)
  Frag16 aq[2][2];
#pragma unroll
  for (int mt = 0; mt < 2; ++mt)
#pragma unroll
    for (int cc = 0; cc < 2; ++cc) {
      const u16* qr = qb + ((size_t)bh * TSEQ + qbase + mt * 16 + l16) * DKH + cc * 32;
#pragma unroll
      for (int v = 0; v < 8; ++v) aq[mt][cc].u[v] = ld32(qr + akp(v, half));
    }

  v8f o[2][4];
  float mrow[2][8], lrow[2][8];
#pragma unroll
  for (int mt = 0; mt < 2; ++mt) {
#pragma unroll
    for (int nt = 0; nt < 4; ++nt) o[mt][nt] = (v8f){0,0,0,0,0,0,0,0};
#pragma unroll
    for (int r = 0; r < 8; ++r) { mrow[mt][r] = -INFINITY; lrow[mt][r] = 0.0f; }
  }

  for (int kb0 = 0; kb0 < qbase + 32; kb0 += 32) {
    // K fragments shared by both q tiles
    Frag16 bk[2][2];   // [key-half][d-chunk]
#pragma unroll
    for (int kh = 0; kh < 2; ++kh)
#pragma unroll
      for (int cc = 0; cc < 2; ++cc) {
        const u16* kr = kbm + ((size_t)bh * TSEQ + kb0 + kh * 16 + l16) * DKH + cc * 32;
#pragma unroll
        for (int v = 0; v < 8; ++v) bk[kh][cc].u[v] = ld32(kr + bkp(v, half));
      }

    const bool needmask = (kb0 + 31 > qbase);
#pragma unroll
    for (int mt = 0; mt < 2; ++mt) {
      v8f s0 = (v8f){0,0,0,0,0,0,0,0}, s1 = s0;
#pragma unroll
      for (int cc = 0; cc < 2; ++cc) {
        s0 = wmma_bf16(aq[mt][cc], bk[0][cc], s0);
        s1 = wmma_bf16(aq[mt][cc], bk[1][cc], s1);
      }
      float alpha[8];
#pragma unroll
      for (int r = 0; r < 8; ++r) {
        float v0 = s0[r], v1 = s1[r];
        if (needmask) {
          int qrow = qbase + mt * 16 + r + 8 * half;
          if (kb0 + l16 > qrow)      v0 = -INFINITY;
          if (kb0 + 16 + l16 > qrow) v1 = -INFINITY;
        }
        float mx = fmaxf(v0, v1);
        mx = fmaxf(mx, __shfl_xor(mx, 1));
        mx = fmaxf(mx, __shfl_xor(mx, 2));
        mx = fmaxf(mx, __shfl_xor(mx, 4));
        mx = fmaxf(mx, __shfl_xor(mx, 8));
        float mnew = fmaxf(mrow[mt][r], mx);
        float al = __expf(mrow[mt][r] - mnew);
        float p0 = __expf(v0 - mnew);
        float p1 = __expf(v1 - mnew);
        float rs = p0 + p1;
        rs += __shfl_xor(rs, 1);
        rs += __shfl_xor(rs, 2);
        rs += __shfl_xor(rs, 4);
        rs += __shfl_xor(rs, 8);
        lrow[mt][r] = lrow[mt][r] * al + rs;
        mrow[mt][r] = mnew;
        alpha[r] = al;
        int row = r + 8 * half;
        plds[wave][mt * 512 + row * 32 + l16]      = f2bf(p0);
        plds[wave][mt * 512 + row * 32 + l16 + 16] = f2bf(p1);
      }
#pragma unroll
      for (int nt = 0; nt < 4; ++nt)
#pragma unroll
        for (int r = 0; r < 8; ++r) o[mt][nt][r] *= alpha[r];
    }

    // wave-local C-layout -> A-layout round trip through LDS
    asm volatile("s_wait_dscnt 0" ::: "memory");
    Frag16 ap[2];
#pragma unroll
    for (int mt = 0; mt < 2; ++mt)
#pragma unroll
      for (int v = 0; v < 8; ++v)
        ap[mt].u[v] = *reinterpret_cast<const u32*>(
            &plds[wave][mt * 512 + l16 * 32 + akp(v, half)]);

#pragma unroll
    for (int nt = 0; nt < 4; ++nt) {
      Frag16 bv;
      const u16* vr = vtb + ((size_t)bh * DKH + nt * 16 + l16) * TSEQ + kb0;
#pragma unroll
      for (int v = 0; v < 8; ++v) bv.u[v] = ld32(vr + bkp(v, half));
#pragma unroll
      for (int mt = 0; mt < 2; ++mt) o[mt][nt] = wmma_bf16(ap[mt], bv, o[mt][nt]);
    }
  }

#pragma unroll
  for (int mt = 0; mt < 2; ++mt)
#pragma unroll
    for (int nt = 0; nt < 4; ++nt)
#pragma unroll
      for (int r = 0; r < 8; ++r) {
        float val = o[mt][nt][r] / lrow[mt][r];
        int q_ = qbase + mt * 16 + r + 8 * half;
        int d = nt * 16 + l16;
        attnb[((size_t)(b * TSEQ + q_)) * DMOD + h * DKH + d] = f2bf(val);
      }
}

// ---------------- output projection ----------------
__global__ void proj_kernel(const u16* __restrict__ ab, const u16* __restrict__ wb,
                            float* __restrict__ out) {
  __shared__ u16 atile[32 * 128];
  const int lane = threadIdx.x & 31, wave = threadIdx.x >> 5;
  const int half = lane >> 4, l16 = lane & 15;
  const int mbase = blockIdx.y * 32;
  const int ebase = blockIdx.x * 512 + wave * 64;

  v8f c[2][4];
  gemm_core(ab, wb, mbase, ebase, DMOD, atile, c);

#pragma unroll
  for (int mt = 0; mt < 2; ++mt)
#pragma unroll
    for (int nt = 0; nt < 4; ++nt)
#pragma unroll
      for (int r = 0; r < 8; ++r) {
        int m = mbase + mt * 16 + r + 8 * half;
        int e = ebase + nt * 16 + l16;
        out[(size_t)m * DMOD + e] = c[mt][nt][r];
      }
}

extern "C" void kernel_launch(void* const* d_in, const int* in_sizes, int n_in,
                              void* d_out, int out_size, void* d_ws, size_t ws_size,
                              hipStream_t stream) {
  const float* x    = (const float*)d_in[0];
  const float* wqkv = (const float*)d_in[1];
  const float* wout = (const float*)d_in[2];
  float* out = (float*)d_out;

  char* ws = (char*)d_ws;
  const size_t SZ_X   = (size_t)BT * DMOD * 2;        // 8 MB
  const size_t SZ_WQ  = (size_t)E3 * DMOD * 2;        // 6 MB
  const size_t SZ_WO  = (size_t)DMOD * DMOD * 2;      // 2 MB
  const size_t SZ_HED = (size_t)NB * NH * TSEQ * DKH * 2; // 8 MB each
  u16* xb   = (u16*)(ws);
  u16* wqb  = (u16*)(ws + SZ_X);
  u16* wob  = (u16*)(ws + SZ_X + SZ_WQ);
  u16* qb   = (u16*)(ws + SZ_X + SZ_WQ + SZ_WO);
  u16* kb   = (u16*)(ws + SZ_X + SZ_WQ + SZ_WO + SZ_HED);
  u16* vtb  = (u16*)(ws + SZ_X + SZ_WQ + SZ_WO + 2 * SZ_HED);
  u16* atb  = (u16*)(ws + SZ_X + SZ_WQ + SZ_WO + 3 * SZ_HED);

  int nx = BT * DMOD, nwq = E3 * DMOD, nwo = DMOD * DMOD;
  cvt_bf16<<<(nx  + 255) / 256, 256, 0, stream>>>(x,    xb,  nx);
  cvt_bf16<<<(nwq + 255) / 256, 256, 0, stream>>>(wqkv, wqb, nwq);
  cvt_bf16<<<(nwo + 255) / 256, 256, 0, stream>>>(wout, wob, nwo);

  qkv_rope_kernel<<<dim3(E3 / 512, BT / 32), 256, 0, stream>>>(xb, wqb, qb, kb, vtb);
  attn_kernel<<<(NB * NH * (TSEQ / 32)) / 8, 256, 0, stream>>>(qb, kb, vtb, atb);
  proj_kernel<<<dim3(DMOD / 512, BT / 32), 256, 0, stream>>>(atb, wob, out);
}
